// CustomLSTM_75754633166960
// MI455X (gfx1250) — matrix operand order repaired and checked by reading, not compiled, not run on hardware
//
#include <hip/hip_runtime.h>
#include <math.h>

// ---------------------------------------------------------------------------
// CustomLSTM for MI455X (gfx1250, wave32, WMMA).
//
// Phase 0: x -> bf16 (row-major), U -> bf16 transposed [n][k]   (BW-bound)
// Phase 1: xW = x_bf @ W_bf + bias    (bf16 WMMA GEMM, 65536x512x2048)
//          A-fragments load directly as b128 from bf16 x — no VALU converts
// Phase 2: persistent recurrence kernel, 32 WGs x 4 waves:
//   - U slice (64KB bf16) resident in LDS for all 1024 steps
//   - cell state c resident in VGPRs for all 1024 steps
//   - h ping-pong (64KB bf16) in global (L2-resident)
//   - global-atomic barrier between steps (monotonic counter)
//   - next-step xW tiles prefetched with global_prefetch_b8
// ---------------------------------------------------------------------------

typedef __attribute__((ext_vector_type(16))) __bf16          bf16x16;
typedef __attribute__((ext_vector_type(8)))  float           f32x8;
typedef __attribute__((ext_vector_type(4)))  float           f32x4;
typedef __attribute__((ext_vector_type(4)))  unsigned int    u32x4;

#define B_   64
#define T_   1024
#define I_   512
#define H_   512
#define G4_  2048          // 4*H
#define LP_  520           // LDS row pitch in bf16 elems (512 + 8 pad -> conflict-free b128)

// float -> bf16 (round to nearest even) — used in convert kernels, W staging and h stores
__device__ __forceinline__ unsigned short f2bf_u(float f) {
    unsigned int u = __builtin_bit_cast(unsigned int, f);
    u += 0x7FFFu + ((u >> 16) & 1u);
    return (unsigned short)(u >> 16);
}

// 16 contiguous bf16 (32B, 16B-aligned) -> WMMA B fragment
__device__ __forceinline__ bf16x16 ld_bf32B(const unsigned short* p) {
    struct Pair { u32x4 lo, hi; } t;
    t.lo = *(const u32x4*)(p);
    t.hi = *(const u32x4*)(p + 8);
    return __builtin_bit_cast(bf16x16, t);
}

// A fragment for k-chunk kc from a row-major bf16 row pointer:
// lane(half,mr): e0..15 -> K = kc*32 + 16*half + e
__device__ __forceinline__ bf16x16 ld_afrag(const unsigned short* row, int kc, int half) {
    struct Pair { u32x4 lo, hi; } t;
    t.lo = *(const u32x4*)(row + kc * 32 + 16 * half);
    t.hi = *(const u32x4*)(row + kc * 32 + 16 * half + 8);
    return __builtin_bit_cast(bf16x16, t);
}

__device__ __forceinline__ float sigf(float x) {
    x = fminf(fmaxf(x, -30.f), 30.f);
    return 1.0f / (1.0f + __expf(-x));
}
__device__ __forceinline__ float tanhfast(float x) {
    x = fminf(fmaxf(x, -15.f), 15.f);
    float e = __expf(-2.0f * x);
    return (1.0f - e) / (1.0f + e);
}

// ------------------------- Phase 0: conversions ----------------------------
// x (B*T*I f32) -> xbf (row-major bf16). 8 elems/thread, fully coalesced.
__global__ __launch_bounds__(256) void cvt_x_kernel(const float* __restrict__ x,
                                                    unsigned short* __restrict__ xbf) {
    size_t i = ((size_t)blockIdx.x * blockDim.x + threadIdx.x) * 8;
    f32x4 a = *(const f32x4*)(x + i);
    f32x4 b = *(const f32x4*)(x + i + 4);
    unsigned short o[8];
#pragma unroll
    for (int e = 0; e < 4; ++e) { o[e] = f2bf_u(a[e]); o[4 + e] = f2bf_u(b[e]); }
    *(u32x4*)(xbf + i) = *(const u32x4*)o;
}

// U (512 k, 2048 n) f32 -> Ubf[n][k] bf16 (2 MB, stays in L2)
__global__ void cvt_u_kernel(const float* __restrict__ U, unsigned short* __restrict__ Ubf) {
    int i = blockIdx.x * blockDim.x + threadIdx.x;
    if (i < I_ * G4_) {
        int k = i / G4_;
        int n = i % G4_;
        Ubf[(size_t)n * H_ + k] = f2bf_u(U[i]);
    }
}

// ------------------------------ state init --------------------------------
__global__ void init_kernel(unsigned short* __restrict__ hbf, unsigned int* __restrict__ ctr) {
    int i = blockIdx.x * blockDim.x + threadIdx.x;
    if (i < 2 * B_ * H_) hbf[i] = 0;
    if (i == 0) *ctr = 0;
}

// --------------------------- Phase 1: xW GEMM ------------------------------
// Block = 256 threads (8 waves). Block tile: 128 rows x 64 cols, K=512 fully.
// W panel staged in LDS as bf16 [n][k] (pitch 520) once per block.
// Inner loop per k-chunk: 2 global b128 (A) + 8 ds b128 (B) + 4 WMMA. No converts.
__global__ __launch_bounds__(256) void gemm_xw_kernel(
    const unsigned short* __restrict__ xbf, const float* __restrict__ W,
    const float* __restrict__ bias, float* __restrict__ xW)
{
    __shared__ unsigned short Wl[64 * LP_];
    const int tid = threadIdx.x;
    const int n0  = blockIdx.x * 64;

    for (int i = tid; i < 64 * 512; i += 256) {
        int k = i >> 6, n = i & 63;
        Wl[n * LP_ + k] = f2bf_u(W[(size_t)k * G4_ + n0 + n]);
    }
    __syncthreads();

    const int lane = tid & 31, wave = tid >> 5;
    const int half = lane >> 4, mr = lane & 15;
    const size_t rowbase = ((size_t)blockIdx.y * 8 + wave) * 16;       // global M base
    const unsigned short* xrow = xbf + (rowbase + mr) * I_;            // A row (bf16)

    f32x8 acc[4];
#pragma unroll
    for (int s = 0; s < 4; ++s) {
        float bv = bias[n0 + s * 16 + mr];
#pragma unroll
        for (int v = 0; v < 8; ++v) acc[s][v] = bv;
    }

#pragma unroll 4
    for (int kc = 0; kc < 16; ++kc) {
        bf16x16 af = ld_afrag(xrow, kc, half);
#pragma unroll
        for (int s = 0; s < 4; ++s) {
            bf16x16 bf = ld_bf32B(&Wl[(s * 16 + mr) * LP_ + kc * 32 + 16 * half]);
            acc[s] = __builtin_amdgcn_wmma_f32_16x16x32_bf16(
                false, af, false, bf, (short)0, acc[s], false, false);
        }
    }

#pragma unroll
    for (int s = 0; s < 4; ++s)
#pragma unroll
        for (int v = 0; v < 8; ++v)
            xW[(rowbase + v + 8 * half) * G4_ + n0 + s * 16 + mr] = acc[s][v];
}

// ------------------------ Phase 2: persistent LSTM -------------------------
// Grid MUST be 32 blocks x 128 threads. Block = hidden n-tile, wave = batch m-tile.
__global__ __launch_bounds__(128) void lstm_rec_kernel(
    const float* __restrict__ xW, const unsigned short* __restrict__ Ubf,
    unsigned short* __restrict__ hbf, float* __restrict__ out,
    unsigned int* __restrict__ ctr, int nwg)
{
    __shared__ unsigned short Ul[64 * LP_];   // [gate*16+nn][k], resident all 1024 steps
    const int tid   = threadIdx.x;
    const int ntile = blockIdx.x;             // 0..31

    for (int i = tid; i < 64 * 512; i += 128) {
        int r = i >> 9, k = i & 511;          // r = gate*16+nn
        int gate = r >> 4, nn = r & 15;
        Ul[r * LP_ + k] = Ubf[(size_t)(gate * 512 + ntile * 16 + nn) * H_ + k];
    }
    __syncthreads();

    const int lane = tid & 31, wave = tid >> 5;   // wave = m_tile (batch rows wave*16..+16)
    const int half = lane >> 4, mr = lane & 15;
    const int m0   = wave * 16;
    const int hcol = ntile * 16 + mr;

    f32x8 c;                                  // cell state lives in VGPRs for all 1024 steps
#pragma unroll
    for (int v = 0; v < 8; ++v) c[v] = 0.0f;

    for (int t = 0; t < T_; ++t) {
        // ---- load A = h_prev tile (bf16, row-major [B][H]) into 16 fragments
        const unsigned short* hc = hbf + (t & 1) * (B_ * H_) + (size_t)(m0 + mr) * H_;
        bf16x16 afr[16];
#pragma unroll
        for (int kc = 0; kc < 16; ++kc) afr[kc] = ld_afrag(hc, kc, half);

        // ---- gates[g] = xW_t tile + h @ U  (4 x 16 WMMAs)
        f32x8 g[4];
#pragma unroll
        for (int gt = 0; gt < 4; ++gt) {
            const float* cb = xW + ((size_t)(m0 + 8 * half) * T_ + t) * G4_ + gt * 512 + hcol;
            f32x8 acc;
#pragma unroll
            for (int v = 0; v < 8; ++v) acc[v] = cb[(size_t)v * T_ * G4_];
            if (t + 1 < T_) {
#pragma unroll
                for (int v = 0; v < 8; ++v)              // hide HBM latency across barrier
                    __builtin_prefetch(cb + G4_ + (size_t)v * T_ * G4_, 0, 0);
            }
#pragma unroll
            for (int kc = 0; kc < 16; ++kc) {
                bf16x16 bf = ld_bf32B(&Ul[(gt * 16 + mr) * LP_ + kc * 32 + 16 * half]);
                acc = __builtin_amdgcn_wmma_f32_16x16x32_bf16(
                    false, afr[kc], false, bf, (short)0, acc, false, false);
            }
            g[gt] = acc;
        }

        // ---- elementwise gate math + stores
        unsigned short* hn = hbf + ((t + 1) & 1) * (B_ * H_);
#pragma unroll
        for (int v = 0; v < 8; ++v) {
            float iv = sigf(g[0][v]);
            float fv = sigf(g[1][v]);
            float gv = tanhfast(g[2][v]);
            float ov = sigf(g[3][v]);
            float cv = fv * c[v] + iv * gv;
            c[v] = cv;
            float hv = ov * tanhfast(cv);
            int b = m0 + v + 8 * half;
            out[((size_t)b * T_ + t) * H_ + hcol] = hv;
            hn[b * H_ + hcol] = f2bf_u(hv);
        }

        // ---- grid-wide barrier (monotonic counter over 32 resident WGs)
        if (t + 1 < T_) {
            __threadfence();
            __syncthreads();
            if (tid == 0) {
                __hip_atomic_fetch_add(ctr, 1u, __ATOMIC_RELEASE, __HIP_MEMORY_SCOPE_AGENT);
                unsigned int tgt = (unsigned int)nwg * (unsigned int)(t + 1);
                while (__hip_atomic_load(ctr, __ATOMIC_ACQUIRE, __HIP_MEMORY_SCOPE_AGENT) < tgt)
                    __builtin_amdgcn_s_sleep(1);
            }
            __syncthreads();
            __builtin_amdgcn_fence(__ATOMIC_ACQUIRE, "agent");
        }
    }
}

// ------------------------------- launcher ----------------------------------
extern "C" void kernel_launch(void* const* d_in, const int* in_sizes, int n_in,
                              void* d_out, int out_size, void* d_ws, size_t ws_size,
                              hipStream_t stream) {
    (void)in_sizes; (void)n_in; (void)out_size;
    const float* x    = (const float*)d_in[0];
    const float* W    = (const float*)d_in[1];
    const float* U    = (const float*)d_in[2];
    const float* bias = (const float*)d_in[3];
    float* out = (float*)d_out;

    const size_t xw_bytes  = (size_t)B_ * T_ * G4_ * sizeof(float);        // 512 MB
    const size_t xbf_bytes = (size_t)B_ * T_ * I_ * sizeof(unsigned short);// 64 MB
    const size_t ubf_bytes = (size_t)G4_ * H_ * sizeof(unsigned short);    // 2 MB
    const size_t hbf_bytes = (size_t)2 * B_ * H_ * sizeof(unsigned short); // 128 KB
    if (ws_size < xw_bytes + xbf_bytes + ubf_bytes + hbf_bytes + 256) return;

    char* ws = (char*)d_ws;
    float*          xW  = (float*)ws;
    unsigned short* xbf = (unsigned short*)(ws + xw_bytes);
    unsigned short* Ubf = (unsigned short*)(ws + xw_bytes + xbf_bytes);
    unsigned short* hbf = (unsigned short*)(ws + xw_bytes + xbf_bytes + ubf_bytes);
    unsigned int*   ctr = (unsigned int*)(ws + xw_bytes + xbf_bytes + ubf_bytes + hbf_bytes);

    cvt_x_kernel<<<(B_ * (size_t)T_ * I_) / (256 * 8), 256, 0, stream>>>(x, xbf);
    cvt_u_kernel<<<(I_ * G4_ + 255) / 256, 256, 0, stream>>>(U, Ubf);
    init_kernel<<<(2 * B_ * H_ + 255) / 256, 256, 0, stream>>>(hbf, ctr);
    gemm_xw_kernel<<<dim3(G4_ / 64, (B_ * T_) / 128), 256, 0, stream>>>(xbf, W, bias, xW);
    lstm_rec_kernel<<<32, 128, 0, stream>>>(xW, Ubf, hbf, out, ctr, 32);
}
